// GlmLayer_24756191494628
// MI455X (gfx1250) — matrix operand order repaired and checked by reading, not compiled, not run on hardware
//
#include <hip/hip_runtime.h>
#include <hip/hip_bf16.h>
#include <math.h>

// ---------------------------------------------------------------------------
// Model constants (match reference)
// ---------------------------------------------------------------------------
namespace {
constexpr int B_    = 2;
constexpr int S_    = 1024;
constexpr int H_    = 1024;
constexpr int NH_   = 8;
constexpr int NOPE_ = 128;
constexpr int VD_   = 128;
constexpr int QHD_  = 192;                       // NOPE + ROPE
constexpr int E_    = 8;
constexpr int INTER_= 1024;
constexpr int SH_   = 1024;
constexpr int T_    = B_ * S_;                   // 2048 tokens
constexpr int QW_   = NH_ * QHD_;                // 1536
constexpr int KVW_  = NH_ * (NOPE_ + VD_) + 64;  // 2112
constexpr int CTXW_ = NH_ * VD_;                 // 1024
constexpr float EPS_   = 1e-6f;
constexpr float SCALE_ = 0.07216878364870323f;   // 192^-0.5
}

typedef __bf16 bf16_t;
typedef __attribute__((ext_vector_type(16))) __bf16 v16bf;
typedef __attribute__((ext_vector_type(8)))  float  v8f;

__device__ __forceinline__ v8f wmma_bf16(v16bf a, v16bf b, v8f c) {
  return __builtin_amdgcn_wmma_f32_16x16x32_bf16(false, a, false, b, (short)0, c,
                                                 false, false);
}

// A 16x32 layout: lanes0-15 hold K=0..15, lanes16-31 hold K=16..31.
__device__ __forceinline__ v16bf frag_a_bf16(const bf16_t* base, int lhi) {
  v16bf a;
#pragma unroll
  for (int j = 0; j < 16; ++j) a[j] = base[lhi * 16 + j];
  return a;
}
// B 32x16 layout: lanes0-15 K=0..15, lanes16-31 K=16..31 (column n per lane).
__device__ __forceinline__ v16bf frag_b_bf16(const bf16_t* base, int lhi) {
  v16bf b;
#pragma unroll
  for (int j = 0; j < 16; ++j) b[j] = base[lhi * 16 + j];
  return b;
}
__device__ __forceinline__ v16bf frag_b_bf16_strided(const bf16_t* base,
                                                     long stride, int lhi) {
  v16bf b;
#pragma unroll
  for (int j = 0; j < 16; ++j) b[j] = base[(long)(lhi * 16 + j) * stride];
  return b;
}

// ---------------------------------------------------------------------------
// Async-LDS double-buffered bf16 GEMM:  C[M,N] = A[M,K] @ W[N,K]^T
// Block 256 threads (8 waves), tile 128x128, K-step 32.
// Tiles stream global->LDS via global_load_async_to_lds_b128 (ASYNCcnt),
// fragments read from LDS, math in v_wmma_f32_16x16x32_bf16.
// EPI: 0 store f32, 1 accum f32, 2 silu->bf16, 3 rowscale-accum f32, 4 bf16
// ---------------------------------------------------------------------------
template <int EPI>
__global__ __launch_bounds__(256) void wmma_gemm_bf16_kernel(
    const bf16_t* __restrict__ A, int lda,
    const bf16_t* __restrict__ W, int ldw,
    void* __restrict__ Cv, int ldc,
    int N, int K,
    const float* __restrict__ rs, int rs_stride) {
  __shared__ __align__(16) bf16_t Atile[2][128][32];  // 16 KB
  __shared__ __align__(16) bf16_t Btile[2][128][32];  // 16 KB

  const int tid  = threadIdx.x;
  const int wid  = tid >> 5;
  const int lane = tid & 31;
  const int lrow = lane & 15;
  const int lhi  = lane >> 4;
  const int mblk = blockIdx.y * 128;
  const int nblk = blockIdx.x * 128;
  const int m0   = (wid & 3) * 32;   // wave sub-tile: 32 rows
  const int n0   = (wid >> 4 ? 0 : (wid >> 2)) * 64;  // wave sub-tile: 64 cols

  // Stage one 128x32 bf16 A tile + B tile: 512 16B chunks each, 256 threads
  // -> 2 chunks of A and 2 of B per thread, issued as async LDS loads.
  auto stage = [&](int buf, int kt) {
#pragma unroll
    for (int s = 0; s < 2; ++s) {
      const int c  = tid + s * 256;
      const int r  = c >> 2;           // row within tile
      const int cc = (c & 3) * 8;      // bf16 column (16B chunk)
      {
        unsigned lds_a = (unsigned)(size_t)(&Atile[buf][r][cc]);
        unsigned long long ga =
            (unsigned long long)(size_t)(A + (size_t)(mblk + r) * lda + kt + cc);
        asm volatile("global_load_async_to_lds_b128 %0, %1, off"
                     :: "v"(lds_a), "v"(ga) : "memory");
      }
      {
        int nr = nblk + r;
        if (nr > N - 1) nr = N - 1;    // clamp: dup row, masked at store
        unsigned lds_b = (unsigned)(size_t)(&Btile[buf][r][cc]);
        unsigned long long gb =
            (unsigned long long)(size_t)(W + (size_t)nr * ldw + kt + cc);
        asm volatile("global_load_async_to_lds_b128 %0, %1, off"
                     :: "v"(lds_b), "v"(gb) : "memory");
      }
    }
  };

  v8f acc[2][4] = {};
  int buf = 0;
  stage(0, 0);
  asm volatile("s_wait_asynccnt 0x0" ::: "memory");
  __syncthreads();

  for (int kt = 0; kt < K; kt += 32) {
    if (kt + 32 < K) stage(buf ^ 1, kt + 32);   // overlap DMA with WMMA

    v16bf a[2], b4[4];
#pragma unroll
    for (int i = 0; i < 2; ++i) {
      const bf16_t* ap = &Atile[buf][m0 + i * 16 + lrow][0];
#pragma unroll
      for (int j = 0; j < 16; ++j) a[i][j] = ap[lhi * 16 + j];
    }
#pragma unroll
    for (int jj = 0; jj < 4; ++jj) {
      const bf16_t* bp = &Btile[buf][n0 + jj * 16 + lrow][lhi * 16];
#pragma unroll
      for (int j = 0; j < 16; ++j) b4[jj][j] = bp[j];
    }
#pragma unroll
    for (int i = 0; i < 2; ++i)
#pragma unroll
      for (int j = 0; j < 4; ++j)
        acc[i][j] = wmma_bf16(a[i], b4[j], acc[i][j]);

    asm volatile("s_wait_asynccnt 0x0" ::: "memory");
    __syncthreads();
    buf ^= 1;
  }

  float*  Cf = (float*)Cv;
  bf16_t* Cb = (bf16_t*)Cv;
#pragma unroll
  for (int i = 0; i < 2; ++i) {
#pragma unroll
    for (int j = 0; j < 4; ++j) {
      const int ncol = nblk + n0 + j * 16 + lrow;
      if (ncol >= N) continue;
#pragma unroll
      for (int r = 0; r < 8; ++r) {
        const int m = mblk + m0 + i * 16 + r + 8 * lhi;
        const float v = acc[i][j][r];
        if (EPI == 0) {
          Cf[(long)m * ldc + ncol] = v;
        } else if (EPI == 1) {
          Cf[(long)m * ldc + ncol] += v;
        } else if (EPI == 2) {
          Cb[(long)m * ldc + ncol] = (bf16_t)(v / (1.f + __expf(-v)));  // SiLU
        } else if (EPI == 3) {
          Cf[(long)m * ldc + ncol] += rs[(long)m * rs_stride] * v;
        } else {
          Cb[(long)m * ldc + ncol] = (bf16_t)v;
        }
      }
    }
  }
}

// ---------------------------------------------------------------------------
// Causal attention on bf16 q/kv: one block per (b, head, 16-query tile).
// Phase 1: scores = Q @ K^T (WMMA) -> bf16 LDS tile; Phase 2: masked softmax;
// Phase 3: O = P @ V (WMMA, strided V fragments) -> bf16 ctx.
// ---------------------------------------------------------------------------
__global__ __launch_bounds__(128) void mla_attn_kernel(
    const bf16_t* __restrict__ q, const bf16_t* __restrict__ kv,
    bf16_t* __restrict__ ctx) {
  const int bh = blockIdx.x;
  const int bb = bh / NH_;
  const int hh = bh % NH_;
  const int q0 = blockIdx.y * 16;

  __shared__ __align__(16) bf16_t sc[16][S_];  // 32 KB score/prob tile
  __shared__ float red[16][10];

  const int tid  = threadIdx.x;  // 128 threads = 4 waves
  const int wid  = tid >> 5;
  const int lane = tid & 31;
  const int lrow = lane & 15;
  const int lhi  = lane >> 4;

  int klen = (q0 + 16 + 31) & ~31;  // causal extent rounded to K-step
  if (klen > S_) klen = S_;

  // ---- Phase 1: Q @ K^T ----
  v16bf aq[4];
  {
    const bf16_t* qbase = q + (long)(bb * S_ + q0 + lrow) * QW_ + hh * QHD_;
#pragma unroll
    for (int kk = 0; kk < 4; ++kk) aq[kk] = frag_a_bf16(qbase + kk * 32, lhi);
  }
  const int c_end = (klen < (wid + 1) * 256) ? klen : (wid + 1) * 256;
  for (int c0 = wid * 256; c0 < c_end; c0 += 16) {
    const bf16_t* kbase = kv + (long)(bb * S_ + c0 + lrow) * KVW_ + hh * NOPE_;
    v8f acc = {};
#pragma unroll
    for (int kk = 0; kk < 4; ++kk)
      acc = wmma_bf16(aq[kk], frag_b_bf16(kbase + kk * 32, lhi), acc);
    const int kcol = c0 + lrow;
#pragma unroll
    for (int r = 0; r < 8; ++r) {
      const int m = r + 8 * lhi;
      const float s = (kcol <= q0 + m) ? acc[r] * SCALE_ : -1e30f;
      sc[m][kcol] = (bf16_t)s;
    }
  }
  __syncthreads();

  // ---- Phase 2: softmax (8 threads per row) ----
  const int row = tid >> 3;
  const int sub = tid & 7;
  float mx = -1e30f;
  for (int c = sub; c < klen; c += 8) mx = fmaxf(mx, (float)sc[row][c]);
  red[row][sub] = mx;
  __syncthreads();
  if (sub == 0) {
    float m2 = red[row][0];
    for (int i = 1; i < 8; ++i) m2 = fmaxf(m2, red[row][i]);
    red[row][8] = m2;
  }
  __syncthreads();
  mx = red[row][8];
  float sum = 0.f;
  for (int c = sub; c < klen; c += 8) sum += __expf((float)sc[row][c] - mx);
  __syncthreads();
  red[row][sub] = sum;
  __syncthreads();
  if (sub == 0) {
    float s2 = 0.f;
    for (int i = 0; i < 8; ++i) s2 += red[row][i];
    red[row][9] = 1.f / s2;
  }
  __syncthreads();
  const float inv = red[row][9];
  for (int c = sub; c < klen; c += 8)
    sc[row][c] = (bf16_t)(__expf((float)sc[row][c] - mx) * inv);
  __syncthreads();

  // ---- Phase 3: O = P @ V (each wave: 32 of the 128 VD columns) ----
  const int n0 = wid * 32;
  v8f o0 = {}, o1 = {};
  for (int kt = 0; kt < klen; kt += 32) {
    v16bf pa;
#pragma unroll
    for (int j = 0; j < 16; ++j) pa[j] = sc[lrow][kt + lhi * 16 + j];
    const bf16_t* vb0 =
        kv + (long)(bb * S_ + kt) * KVW_ + NH_ * NOPE_ + hh * VD_ + n0 + lrow;
    o0 = wmma_bf16(pa, frag_b_bf16_strided(vb0, KVW_, lhi), o0);
    o1 = wmma_bf16(pa, frag_b_bf16_strided(vb0 + 16, KVW_, lhi), o1);
  }
  bf16_t* cb = ctx + (long)(bb * S_ + q0) * CTXW_ + hh * VD_;
#pragma unroll
  for (int r = 0; r < 8; ++r) {
    const int m = r + 8 * lhi;
    cb[(long)m * CTXW_ + n0 + lrow]      = (bf16_t)o0[r];
    cb[(long)m * CTXW_ + n0 + 16 + lrow] = (bf16_t)o1[r];
  }
}

// ---------------------------------------------------------------------------
// RMSNorm (f32 in -> bf16 out), one block per token row, wave32 reduce
// ---------------------------------------------------------------------------
__global__ __launch_bounds__(256) void rmsnorm_kernel(
    const float* __restrict__ x, const float* __restrict__ w,
    bf16_t* __restrict__ out) {
  const int row = blockIdx.x;
  const float* xr = x + (long)row * H_;
  bf16_t* orow = out + (long)row * H_;
  __shared__ float red[8];
  __shared__ float s_inv;
  const int tid = threadIdx.x;
  float s = 0.f;
  for (int i = tid; i < H_; i += 256) { const float v = xr[i]; s += v * v; }
#pragma unroll
  for (int off = 16; off > 0; off >>= 1) s += __shfl_down(s, off, 32);
  if ((tid & 31) == 0) red[tid >> 5] = s;
  __syncthreads();
  if (tid == 0) {
    float t = 0.f;
    for (int i = 0; i < 8; ++i) t += red[i];
    s_inv = rsqrtf(t / (float)H_ + EPS_);
  }
  __syncthreads();
  const float inv = s_inv;
  for (int i = tid; i < H_; i += 256) orow[i] = (bf16_t)(xr[i] * inv * w[i]);
}

// ---------------------------------------------------------------------------
// Router: softmax over 8 experts, top-2, renormalize -> dense [T,E] weights
// ---------------------------------------------------------------------------
__global__ void gate_kernel(const bf16_t* __restrict__ h2,
                            const float* __restrict__ gw,
                            float* __restrict__ wfull) {
  const int t = blockIdx.x * blockDim.x + threadIdx.x;
  if (t >= T_) return;
  const bf16_t* hr = h2 + (long)t * H_;
  float p[E_];
  float mx = -1e30f;
  for (int e = 0; e < E_; ++e) {
    const float* g = gw + (long)e * H_;
    float d = 0.f;
    for (int i = 0; i < H_; ++i) d += (float)hr[i] * g[i];
    p[e] = d;
    mx = fmaxf(mx, d);
  }
  float sum = 0.f;
  for (int e = 0; e < E_; ++e) { p[e] = __expf(p[e] - mx); sum += p[e]; }
  for (int e = 0; e < E_; ++e) p[e] /= sum;
  int i1 = 0;
  for (int e = 1; e < E_; ++e) if (p[e] > p[i1]) i1 = e;
  int i2 = (i1 == 0) ? 1 : 0;
  for (int e = 0; e < E_; ++e) if (e != i1 && p[e] > p[i2]) i2 = e;
  const float s2 = p[i1] + p[i2];
  for (int e = 0; e < E_; ++e) wfull[(long)t * E_ + e] = 0.f;
  wfull[(long)t * E_ + i1] = p[i1] / s2;
  wfull[(long)t * E_ + i2] = p[i2] / s2;
}

// ---------------------------------------------------------------------------
// Elementwise helpers
// ---------------------------------------------------------------------------
__global__ void cvt_bf16_kernel(const float* __restrict__ a,
                                bf16_t* __restrict__ o, int n) {
  const int i = blockIdx.x * blockDim.x + threadIdx.x;
  if (i < n) o[i] = (bf16_t)a[i];
}
__global__ void copy_kernel(const float* __restrict__ a, float* __restrict__ o,
                            int n) {
  const int i = blockIdx.x * blockDim.x + threadIdx.x;
  if (i < n) o[i] = a[i];
}
__global__ void zero_kernel(float* __restrict__ o, int n) {
  const int i = blockIdx.x * blockDim.x + threadIdx.x;
  if (i < n) o[i] = 0.f;
}
__global__ void add_kernel(const float* __restrict__ a,
                           const float* __restrict__ b, float* __restrict__ o,
                           int n) {
  const int i = blockIdx.x * blockDim.x + threadIdx.x;
  if (i < n) o[i] = a[i] + b[i];
}
__global__ void silumul_kernel(const float* __restrict__ gu,
                               bf16_t* __restrict__ o, int n) {
  const int i = blockIdx.x * blockDim.x + threadIdx.x;
  if (i < n) {
    const int t = i / SH_;
    const int c = i % SH_;
    const float g = gu[(long)t * (2 * SH_) + c];
    const float u = gu[(long)t * (2 * SH_) + SH_ + c];
    o[i] = (bf16_t)(g / (1.f + __expf(-g)) * u);
  }
}

// ---------------------------------------------------------------------------
// Orchestration
// ---------------------------------------------------------------------------
extern "C" void kernel_launch(void* const* d_in, const int* in_sizes, int n_in,
                              void* d_out, int out_size, void* d_ws, size_t ws_size,
                              hipStream_t stream) {
  (void)in_sizes; (void)n_in; (void)out_size; (void)ws_size;
  const float* x     = (const float*)d_in[0];
  // d_in[1] positions: unused (reference drops rope dims in this path)
  const float* ln1   = (const float*)d_in[2];
  const float* ln2   = (const float*)d_in[3];
  const float* Wq    = (const float*)d_in[4];
  const float* Wkv   = (const float*)d_in[5];
  const float* Wo    = (const float*)d_in[6];
  const float* gw    = (const float*)d_in[7];
  const float* w1    = (const float*)d_in[8];
  const float* w2    = (const float*)d_in[9];
  const float* sgu   = (const float*)d_in[10];
  const float* sdown = (const float*)d_in[11];
  float* out = (float*)d_out;

  char* ws = (char*)d_ws;
  size_t off = 0;
  auto carve = [&](size_t bytes) {
    char* p = ws + off;
    off = (off + bytes + 255) & ~(size_t)255;
    return p;
  };
  // bf16 weight copies (converted once per launch)
  bf16_t* wqB  = (bf16_t*)carve((size_t)QW_ * H_ * 2);
  bf16_t* wkvB = (bf16_t*)carve((size_t)KVW_ * H_ * 2);
  bf16_t* woB  = (bf16_t*)carve((size_t)H_ * CTXW_ * 2);
  bf16_t* w1B  = (bf16_t*)carve((size_t)E_ * INTER_ * H_ * 2);
  bf16_t* w2B  = (bf16_t*)carve((size_t)E_ * H_ * INTER_ * 2);
  bf16_t* sguB = (bf16_t*)carve((size_t)2 * SH_ * H_ * 2);
  bf16_t* sdB  = (bf16_t*)carve((size_t)H_ * SH_ * 2);
  // activations
  bf16_t* hB   = (bf16_t*)carve((size_t)T_ * H_ * 2);     // h1 / h2 (bf16)
  bf16_t* qB   = (bf16_t*)carve((size_t)T_ * QW_ * 2);    // q -> reused as he
  bf16_t* kvB  = (bf16_t*)carve((size_t)T_ * KVW_ * 2);   // kv
  bf16_t* ctxB = (bf16_t*)carve((size_t)T_ * CTXW_ * 2);  // ctx -> reused sact
  float*  gu   = (float*)carve((size_t)T_ * 2 * SH_ * 4); // shared gate_up
  float*  x1   = (float*)carve((size_t)T_ * H_ * 4);      // x + attn
  float*  rt   = (float*)carve((size_t)T_ * H_ * 4);      // routed accumulator
  float*  wf   = (float*)carve((size_t)T_ * E_ * 4);      // router weights

  auto cvt = [&](const float* src, bf16_t* dst, long n) {
    cvt_bf16_kernel<<<(int)((n + 255) / 256), 256, 0, stream>>>(src, dst, (int)n);
  };
  // 0. one-shot weight conversion to bf16 (bandwidth-trivial, saves 16x
  //    redundant per-tile cvt work inside the GEMMs)
  cvt(Wq, wqB, (long)QW_ * H_);
  cvt(Wkv, wkvB, (long)KVW_ * H_);
  cvt(Wo, woB, (long)H_ * CTXW_);
  cvt(w1, w1B, (long)E_ * INTER_ * H_);
  cvt(w2, w2B, (long)E_ * H_ * INTER_);
  cvt(sgu, sguB, (long)2 * SH_ * H_);
  cvt(sdown, sdB, (long)H_ * SH_);

  // 1. h1 = rmsnorm(x, ln1) -> bf16
  rmsnorm_kernel<<<T_, 256, 0, stream>>>(x, ln1, hB);
  // 2. q = h1 @ Wq^T (bf16 out)
  wmma_gemm_bf16_kernel<4><<<dim3((QW_ + 127) / 128, T_ / 128), 256, 0, stream>>>(
      hB, H_, wqB, H_, qB, QW_, QW_, H_, nullptr, 0);
  // 3. kv = h1 @ Wkv^T (bf16 out)
  wmma_gemm_bf16_kernel<4><<<dim3((KVW_ + 127) / 128, T_ / 128), 256, 0, stream>>>(
      hB, H_, wkvB, H_, kvB, KVW_, KVW_, H_, nullptr, 0);
  // 4. causal attention -> ctx (bf16)
  mla_attn_kernel<<<dim3(B_ * NH_, S_ / 16), 128, 0, stream>>>(qB, kvB, ctxB);
  // 5. x1 = x + ctx @ Wo^T
  copy_kernel<<<(T_ * H_ + 255) / 256, 256, 0, stream>>>(x, x1, T_ * H_);
  wmma_gemm_bf16_kernel<1><<<dim3(H_ / 128, T_ / 128), 256, 0, stream>>>(
      ctxB, CTXW_, woB, CTXW_, x1, H_, H_, CTXW_, nullptr, 0);
  // 6. h2 = rmsnorm(x1, ln2) -> bf16
  rmsnorm_kernel<<<T_, 256, 0, stream>>>(x1, ln2, hB);
  // 7. router weights
  gate_kernel<<<(T_ + 255) / 256, 256, 0, stream>>>(hB, gw, wf);
  // 8. routed = 0
  zero_kernel<<<(T_ * H_ + 255) / 256, 256, 0, stream>>>(rt, T_ * H_);
  // 9. experts (dense, matching reference): he = silu(h2 @ w1[e]^T) (bf16);
  //    routed += wfull[:,e] * (he @ w2[e]^T)
  bf16_t* heB = qB;  // reuse
  for (int e = 0; e < E_; ++e) {
    wmma_gemm_bf16_kernel<2><<<dim3(INTER_ / 128, T_ / 128), 256, 0, stream>>>(
        hB, H_, w1B + (size_t)e * INTER_ * H_, H_, heB, INTER_, INTER_, H_,
        nullptr, 0);
    wmma_gemm_bf16_kernel<3><<<dim3(H_ / 128, T_ / 128), 256, 0, stream>>>(
        heB, INTER_, w2B + (size_t)e * H_ * INTER_, INTER_, rt, H_, H_, INTER_,
        wf + e, E_);
  }
  // 10. shared expert
  wmma_gemm_bf16_kernel<0><<<dim3((2 * SH_) / 128, T_ / 128), 256, 0, stream>>>(
      hB, H_, sguB, H_, gu, 2 * SH_, 2 * SH_, H_, nullptr, 0);
  bf16_t* saB = ctxB;  // reuse
  silumul_kernel<<<(T_ * SH_ + 255) / 256, 256, 0, stream>>>(gu, saB, T_ * SH_);
  wmma_gemm_bf16_kernel<1><<<dim3(H_ / 128, T_ / 128), 256, 0, stream>>>(
      saB, SH_, sdB, SH_, rt, H_, H_, SH_, nullptr, 0);
  // 11. out = x1 + routed
  add_kernel<<<(T_ * H_ + 255) / 256, 256, 0, stream>>>(x1, rt, out, T_ * H_);
}
